// MultiHeadAttention_14508399526744
// MI455X (gfx1250) — matrix operand order repaired and checked by reading, not compiled, not run on hardware
//
#include <hip/hip_runtime.h>
#include <hip/hip_bf16.h>

typedef __attribute__((ext_vector_type(16))) _Float16 v16h;
typedef __attribute__((ext_vector_type(8)))  _Float16 v8h;
typedef __attribute__((ext_vector_type(8)))  float    v8f;

#define WMMA_F32_F16(a, b, c) \
  __builtin_amdgcn_wmma_f32_16x16x32_f16(false, (a), false, (b), (short)0, (c), false, false)

constexpr int Bc  = 16;
constexpr int Sc  = 784;
constexpr int Dc  = 1024;
constexpr int Hc  = 16;
constexpr int DKc = 64;
constexpr int ROWTILES  = (Bc * Sc) / 16;  // 784 tiles of 16 rows over B*S
constexpr int QT_PER_BH = Sc / 16;         // 49

// ---------------------------------------------------------------------------
// Kernel 0: one-shot fp32 -> f16 conversion of the four DxD weight matrices.
// Each thread converts 8 contiguous elements (2x float4 in, 1x v8h out).
// ---------------------------------------------------------------------------
__global__ __launch_bounds__(256) void wcvt_kernel(
    const float* __restrict__ Wq, const float* __restrict__ Wk,
    const float* __restrict__ Wv, const float* __restrict__ Wo,
    _Float16* __restrict__ Wh)   // 4 matrices back-to-back, each Dc*Dc
{
  const long idx = ((long)blockIdx.x * 256 + threadIdx.x) * 8;   // over 4*Dc*Dc
  const int  mat = (int)(idx >> 20);                             // Dc*Dc = 1<<20
  const long off = idx & ((1l << 20) - 1);
  const float* src = (mat == 0) ? Wq : (mat == 1) ? Wk : (mat == 2) ? Wv : Wo;
  const float4* s = (const float4*)(src + off);
  const float4 a = s[0], b = s[1];
  v8h r;
  r[0] = (_Float16)a.x; r[1] = (_Float16)a.y; r[2] = (_Float16)a.z; r[3] = (_Float16)a.w;
  r[4] = (_Float16)b.x; r[5] = (_Float16)b.y; r[6] = (_Float16)b.z; r[7] = (_Float16)b.w;
  *(v8h*)(Wh + idx) = r;
}

// ---------------------------------------------------------------------------
// Kernel 1: fused QKV projection (f16 WMMA, f32 accum) + bias + 2D RoPE.
//   q,k -> [B,H,S,64] f16 (RoPE applied);  v -> [B,H,64,S] f16 (transposed).
// One wave computes a 16x64 output tile; K loop of 32 over D=1024.
// Weights are pre-converted f16 -> B fragment is one aligned v16h load.
// ---------------------------------------------------------------------------
__global__ __launch_bounds__(256) void proj_rope_kernel(
    const float* __restrict__ Xq, const float* __restrict__ Xk, const float* __restrict__ Xv,
    const _Float16* __restrict__ Wh,   // [3][Dc][Dc] f16 (q,k,v)
    const float* __restrict__ bq, const float* __restrict__ bk, const float* __restrict__ bvp,
    _Float16* __restrict__ qbuf, _Float16* __restrict__ kbuf, _Float16* __restrict__ vTbuf)
{
  const int lane = threadIdx.x & 31;
  const int wid  = threadIdx.x >> 5;
  const int w    = blockIdx.x * 8 + wid;          // 0 .. 3*784*16-1
  const int mat  = w / (ROWTILES * 16);           // 0=q 1=k 2=v
  const int rem  = w % (ROWTILES * 16);
  const int rowTile = rem >> 4;                   // 0..783
  const int colBase = (rem & 15) * 64;
  const int l16 = lane & 15;
  const int hi  = lane >> 4;

  const float*    X    = (mat == 0) ? Xq : (mat == 1) ? Xk : Xv;
  const _Float16* W    = Wh + (long)mat * Dc * Dc;
  const float*    bias = (mat == 0) ? bq : (mat == 1) ? bk : bvp;

  const float* arow = X + (long)(rowTile * 16 + l16) * Dc;   // A: lane = M

  v8f acc[4] = {};
  for (int kb = 0; kb < Dc; kb += 32) {
    // A fragment 16x32: elems 0..15 -> K = kb + 16*hi + e
    v16h a;
    const int ka0 = kb + hi * 16;
    const int ka1 = kb + hi * 16 + 8;
#pragma unroll
    for (int e = 0; e < 8; ++e) {
      a[e]     = (_Float16)arow[ka0 + e];
      a[e + 8] = (_Float16)arow[ka1 + e];
    }
    // B fragment 32x16: B[k,n] = W[n,k]; lane = N, elems K = kb + 16*hi + e
    const int kbm = kb + hi * 16;
#pragma unroll
    for (int sub = 0; sub < 4; ++sub) {
      const v16h bf = *(const v16h*)(W + (long)(colBase + sub * 16 + l16) * Dc + kbm);
      acc[sub] = WMMA_F32_F16(a, bf, acc[sub]);
    }
  }

  // Epilogue: bias + RoPE (q,k) + scatter to attention-friendly layouts.
  const int bb    = rowTile / QT_PER_BH;          // batch
  const int sBase = (rowTile % QT_PER_BH) * 16;   // sequence base
#pragma unroll
  for (int sub = 0; sub < 4; ++sub) {
    const int n = colBase + sub * 16 + l16;       // C: lane = N
    const float biasn = bias[n];
    const int h = n >> 6, d = n & 63;
    float vals[8];
#pragma unroll
    for (int i = 0; i < 8; ++i) vals[i] = acc[sub][i] + biasn;

    if (mat < 2) {  // 2D RoPE: combined angle (r + c) * theta_j, pair across lanes n^1
      const float theta = __expf(-(float)(d & ~1) * (9.210340371976184f / 64.0f)); // BASE^-(2j/64)
      const bool even = (d & 1) == 0;
#pragma unroll
      for (int i = 0; i < 8; ++i) {
        const int s = sBase + i + hi * 8;         // C: VGPR i -> M = i + 8*hi
        const int r = s / 28, c = s - r * 28;
        float sn, cs;
        __sincosf((float)(r + c) * theta, &sn, &cs);
        const float x = vals[i];
        const float p = __shfl_xor(x, 1, 32);     // partner feature of the rotation pair
        vals[i] = even ? (x * cs - p * sn) : (x * cs + p * sn);
      }
    }

    if (mat == 2) {   // v transposed: [B,H,64,S]
      _Float16* dst = vTbuf + ((long)(bb * Hc + h) * DKc + d) * Sc;
#pragma unroll
      for (int i = 0; i < 8; ++i) dst[sBase + i + hi * 8] = (_Float16)vals[i];
    } else {          // q/k: [B,H,S,64]
      _Float16* dst = ((mat == 0) ? qbuf : kbuf) + (long)(bb * Hc + h) * Sc * DKc + d;
#pragma unroll
      for (int i = 0; i < 8; ++i) dst[(long)(sBase + i + hi * 8) * DKc] = (_Float16)vals[i];
    }
  }
}

// ---------------------------------------------------------------------------
// Kernel 2: attention. One wave = 16 query rows of one (b,h).
//   scores = q k^T / 8 (WMMA), softmax over S=784 in LDS, out = P @ v (WMMA).
// ---------------------------------------------------------------------------
__global__ __launch_bounds__(64) void attn_kernel(
    const _Float16* __restrict__ qbuf, const _Float16* __restrict__ kbuf,
    const _Float16* __restrict__ vTbuf, _Float16* __restrict__ obuf)
{
  __shared__ float sc[2][16][Sc];                 // 100,352 B (2 waves/block)
  const int lane = threadIdx.x & 31;
  const int wl   = threadIdx.x >> 5;
  const int w    = blockIdx.x * 2 + wl;           // 0 .. B*H*49-1
  const int bh   = w / QT_PER_BH;
  const int qt   = w % QT_PER_BH;
  const int l16  = lane & 15, hi = lane >> 4;

  // Hold both Q A-fragments (d=0..31, 32..63) in registers for the whole pass.
  const v8h* qv = (const v8h*)(qbuf + ((long)bh * Sc + qt * 16 + l16) * DKc);
  const v8h c0 = qv[2 * hi], c1 = qv[2 * hi + 1], c2 = qv[4 + 2 * hi], c3 = qv[5 + 2 * hi];
  v16h aq0, aq1;
#pragma unroll
  for (int e = 0; e < 8; ++e) { aq0[e] = c0[e]; aq0[e+8] = c1[e]; aq1[e] = c2[e]; aq1[e+8] = c3[e]; }

  // Pass 1: scores -> LDS
  const _Float16* kbase = kbuf + (long)bh * Sc * DKc;
  for (int kt = 0; kt < QT_PER_BH; ++kt) {
    const v16h* kv = (const v16h*)(kbase + (long)(kt * 16 + l16) * DKc);  // B: lane = key n
    const v16h bk0 = kv[hi];        // K = d in [0,32)
    const v16h bk1 = kv[2 + hi];    // K = d in [32,64)
    v8f s = {};
    s = WMMA_F32_F16(aq0, bk0, s);
    s = WMMA_F32_F16(aq1, bk1, s);
#pragma unroll
    for (int i = 0; i < 8; ++i) sc[wl][i + hi * 8][kt * 16 + l16] = s[i] * 0.125f; // 1/sqrt(64)
  }

  // Softmax: 2 lanes per row, each over 392 columns, shfl-combined.
  {
    float* prow = sc[wl][lane >> 1];
    const int cb = (lane & 1) * (Sc / 2);
    float mx = -3.0e38f;
    for (int c = 0; c < Sc / 2; ++c) mx = fmaxf(mx, prow[cb + c]);
    mx = fmaxf(mx, __shfl_xor(mx, 1, 32));
    float sum = 0.f;
    for (int c = 0; c < Sc / 2; ++c) { float e = __expf(prow[cb + c] - mx); prow[cb + c] = e; sum += e; }
    sum += __shfl_xor(sum, 1, 32);
    const float inv = 1.0f / sum;
    for (int c = 0; c < Sc / 2; ++c) prow[cb + c] *= inv;
  }

  // Pass 2: out = P @ v, K = s in steps of 32 (784 = 24*32 + 16 tail, zero-padded)
  const _Float16* vb = vTbuf + (long)bh * DKc * Sc;
  v8f o[4] = {};
  for (int base = 0; base < Sc; base += 32) {
    v16h ap;
#pragma unroll
    for (int e = 0; e < 8; ++e) {
      const int s0 = base + hi * 16 + e;
      const int s1 = base + hi * 16 + 8 + e;
      ap[e]     = (s0 < Sc) ? (_Float16)sc[wl][l16][s0] : (_Float16)0.f;
      ap[e + 8] = (s1 < Sc) ? (_Float16)sc[wl][l16][s1] : (_Float16)0.f;
    }
#pragma unroll
    for (int sub = 0; sub < 4; ++sub) {
      const _Float16* vrow = vb + (long)(sub * 16 + l16) * Sc;  // B: lane = N = d
      const int ks = base + hi * 16;
      v16h bv;
      if (ks + 15 < Sc) {
        bv = *(const v16h*)(vrow + ks);            // contiguous 32B, 32B-aligned
      } else {
#pragma unroll
        for (int e = 0; e < 16; ++e) bv[e] = (ks + e < Sc) ? vrow[ks + e] : (_Float16)0.f;
      }
      o[sub] = WMMA_F32_F16(ap, bv, o[sub]);
    }
  }

  // Store attention output as [B,S,D] f16, ready as A for the output projection.
  const int b = bh >> 4, h = bh & 15;
#pragma unroll
  for (int sub = 0; sub < 4; ++sub) {
    const int d = sub * 16 + l16;
#pragma unroll
    for (int i = 0; i < 8; ++i) {
      const int srow = qt * 16 + i + hi * 8;
      obuf[(long)(b * Sc + srow) * Dc + h * DKc + d] = (_Float16)o[sub][i];
    }
  }
}

// ---------------------------------------------------------------------------
// Kernel 3: output projection  out = obuf @ W_o^T + b_o  (fp32 result).
// A (obuf) and B (pre-converted W_o) are both f16: no conversions in the loop.
// ---------------------------------------------------------------------------
__global__ __launch_bounds__(256) void out_proj_kernel(
    const _Float16* __restrict__ obuf, const _Float16* __restrict__ Woh,
    const float* __restrict__ bo, float* __restrict__ out)
{
  const int lane = threadIdx.x & 31;
  const int wid  = threadIdx.x >> 5;
  const int w    = blockIdx.x * 8 + wid;          // 0..12543
  const int rowTile = w >> 4;
  const int colBase = (w & 15) * 64;
  const int l16 = lane & 15, hi = lane >> 4;

  const v8h* av = (const v8h*)(obuf + (long)(rowTile * 16 + l16) * Dc);
  v8f acc[4] = {};
  for (int kc = 0; kc < Dc / 8; kc += 4) {        // K step = 32 elems = 4 chunks of 8
    const v8h a0 = av[kc + 2 * hi], a1 = av[kc + 2 * hi + 1];
    v16h a;
#pragma unroll
    for (int e = 0; e < 8; ++e) { a[e] = a0[e]; a[e + 8] = a1[e]; }
    const int kbm = kc * 8 + hi * 16;
#pragma unroll
    for (int sub = 0; sub < 4; ++sub) {
      const v16h bf = *(const v16h*)(Woh + (long)(colBase + sub * 16 + l16) * Dc + kbm);
      acc[sub] = WMMA_F32_F16(a, bf, acc[sub]);
    }
  }
#pragma unroll
  for (int sub = 0; sub < 4; ++sub) {
    const int n = colBase + sub * 16 + l16;
    const float biasn = bo[n];
#pragma unroll
    for (int i = 0; i < 8; ++i)
      out[(long)(rowTile * 16 + i + hi * 8) * Dc + n] = acc[sub][i] + biasn;
  }
}

// ---------------------------------------------------------------------------
extern "C" void kernel_launch(void* const* d_in, const int* in_sizes, int n_in,
                              void* d_out, int out_size, void* d_ws, size_t ws_size,
                              hipStream_t stream) {
  const float* Q  = (const float*)d_in[0];
  const float* K  = (const float*)d_in[1];
  const float* V  = (const float*)d_in[2];
  const float* Wq = (const float*)d_in[3];
  const float* bq = (const float*)d_in[4];
  const float* Wk = (const float*)d_in[5];
  const float* bk = (const float*)d_in[6];
  const float* Wv = (const float*)d_in[7];
  const float* bv = (const float*)d_in[8];
  const float* Wo = (const float*)d_in[9];
  const float* bo = (const float*)d_in[10];
  // d_in[11] = grid_sizes (always 28x28; baked in)
  float* out = (float*)d_out;

  const size_t elems = (size_t)Bc * Sc * Dc;      // 12,845,056 f16 elems per act buffer
  const size_t welems = (size_t)Dc * Dc;          // 1,048,576 f16 elems per weight
  _Float16* qbuf = (_Float16*)d_ws;
  _Float16* kbuf = qbuf + elems;
  _Float16* vT   = kbuf + elems;
  _Float16* obuf = vT + elems;
  _Float16* Wh   = obuf + elems;                  // [4][Dc][Dc]: Wq,Wk,Wv,Wo
  _Float16* Woh  = Wh + 3 * welems;

  wcvt_kernel<<<(4 * welems) / (256 * 8), 256, 0, stream>>>(Wq, Wk, Wv, Wo, Wh);
  proj_rope_kernel<<<(3 * ROWTILES * 16) / 8, 256, 0, stream>>>(
      Q, K, V, Wh, bq, bk, bv, qbuf, kbuf, vT);
  attn_kernel<<<(Bc * Hc * QT_PER_BH) / 2, 64, 0, stream>>>(qbuf, kbuf, vT, obuf);
  out_proj_kernel<<<(ROWTILES * 16) / 8, 256, 0, stream>>>(obuf, Woh, bo, out);
}
